// DecoderLayer_65266323030558
// MI455X (gfx1250) — matrix operand order repaired and checked by reading, not compiled, not run on hardware
//
#include <hip/hip_runtime.h>
#include <hip/hip_bf16.h>
#include <math.h>

typedef __bf16 bf16;
typedef __attribute__((ext_vector_type(16))) __bf16 v16bf;
typedef __attribute__((ext_vector_type(8)))  float  v8f;

#define B_   4
#define T_   512
#define D_   512
#define H_   8
#define DH_  64
#define E_   24
#define HID_ 2048
#define N_   (B_ * T_)   // 2048 tokens
#define BH_  (B_ * H_)   // 32 attention batches
#define NP_  (N_ * 2)    // 4096 (token, expert) pairs

union FragU { v16bf v; uint4 q[2]; };

// ---------------------------------------------------------------------------
// Generic 128x128x32 bf16-WMMA GEMM.
//   AFMT   : 0 = A is fp32 (convert on LDS fill), 1 = A already bf16
//   BTRANS : 0 = B row-major [Kd,Nd], 1 = B element (k,n) = Bsrc[n*ldb + k]
//   CAUSAL : mask col>row with -1e30 (attention scores)
//   EPI    : 0 = C = scale*acc, 1 = C = scale*acc + Res
//   GROUPED: MoE grouped mode: M per batch from m_counts, rows via group_offs
//            (+ optional a_rowmap token gather); C row = global pair index.
//   CT     : float or bf16 output
// Batch offset = (z/innerCnt)*outerStride + (z%innerCnt)*innerStride.
// ---------------------------------------------------------------------------
template <int AFMT, int BTRANS, int CAUSAL, int EPI, int GROUPED, typename CT>
__global__ __launch_bounds__(256) void gemm_wmma(
    const void* __restrict__ Av, const float* __restrict__ Bsrc,
    CT* __restrict__ C, const float* __restrict__ Res,
    int M, int Nd, int Kd, int lda, int ldb, int ldc,
    long long aOut, long long aIn, long long bOut, long long bIn,
    long long cOut, long long cIn, int innerCnt, float scale,
    const int* __restrict__ m_counts, const int* __restrict__ group_offs,
    const int* __restrict__ a_rowmap) {
  __shared__ bf16 ldsA[128 * 40];  // padded stride 40 -> conflict-free b128 frags
  __shared__ bf16 ldsB[128 * 40];  // stored transposed: ldsB[n][k]

  const int tid = threadIdx.x;
  const int lane = tid & 31;
  const int wid = tid >> 5;
  const int wm = (wid >> 2) * 64;  // wave tile 64x32 in block tile 128x128
  const int wn = (wid & 3) * 32;
  const int bz = blockIdx.z;
  const int rowbase = blockIdx.y * 128;
  const int colbase = blockIdx.x * 128;

  const long long aoff = (long long)(bz / innerCnt) * aOut + (long long)(bz % innerCnt) * aIn;
  const long long boff = (long long)(bz / innerCnt) * bOut + (long long)(bz % innerCnt) * bIn;
  const long long coff = (long long)(bz / innerCnt) * cOut + (long long)(bz % innerCnt) * cIn;

  int Mloc = M;
  int goff = 0;
  if (GROUPED) {
    Mloc = m_counts[bz];
    goff = group_offs[bz];
    if (rowbase >= Mloc) return;  // uniform across block
  }

  const float* Af = (const float*)Av;
  const bf16*  Ab = (const bf16*)Av;

  v8f acc[4][2];
  for (int mi = 0; mi < 4; ++mi)
    for (int ni = 0; ni < 2; ++ni)
      for (int r = 0; r < 8; ++r) acc[mi][ni][r] = 0.0f;

  for (int kk = 0; kk < Kd; kk += 32) {
    if (kk + 32 < Kd)  // gfx1250 global_prefetch on next weight tile
      __builtin_prefetch((const void*)(Bsrc + boff + (long long)(kk + 32) * ldb), 0, 1);
    __syncthreads();

    // ---- A tile: 128 rows x 32 cols bf16, 8 elements (16B) per thread-step
    for (int it = 0; it < 2; ++it) {
      int lin = tid + it * 256;
      int r = lin >> 2;
      int kc = (lin & 3) << 3;
      int lr = rowbase + r;
      union { uint4 q; bf16 h[8]; } t;
      t.q.x = t.q.y = t.q.z = t.q.w = 0u;
      if (lr < Mloc) {
        long long arow;
        if (GROUPED) {
          int g = goff + lr;
          arow = a_rowmap ? (long long)a_rowmap[g] : (long long)g;
        } else {
          arow = lr;
        }
        if (AFMT == 0) {
          const float* p = Af + aoff + arow * (long long)lda + (kk + kc);
          for (int j = 0; j < 8; ++j) t.h[j] = (bf16)p[j];
        } else {
          t.q = *(const uint4*)(Ab + aoff + arow * (long long)lda + (kk + kc));
        }
      }
      *(uint4*)(&ldsA[r * 40 + kc]) = t.q;
    }

    // ---- B tile: fp32 weights -> bf16, stored transposed ldsB[n][k]
    if (BTRANS == 0) {
      int kb = tid >> 5;          // 0..7
      int n4 = (tid & 31) << 2;   // 0..124
      for (int p = 0; p < 4; ++p) {
        int k = kb + p * 8;
        for (int j = 0; j < 4; ++j) {
          int gn = colbase + n4 + j;
          float v = 0.0f;
          if (gn < Nd) v = Bsrc[boff + (long long)(kk + k) * ldb + gn];
          ldsB[(n4 + j) * 40 + k] = (bf16)v;
        }
      }
    } else {
      int nb = tid >> 3;          // 0..31
      int kq = (tid & 7) << 2;    // 0..28
      for (int p = 0; p < 4; ++p) {
        int n = nb + p * 32;
        int gn = colbase + n;
        bool nv = gn < Nd;
        for (int j = 0; j < 4; ++j) {
          float v = 0.0f;
          if (nv) v = Bsrc[boff + (long long)gn * ldb + (kk + kq + j)];
          ldsB[n * 40 + kq + j] = (bf16)v;
        }
      }
    }
    __syncthreads();

    // ---- fragments (documented CDNA5 16-bit A / B VGPR layouts) + WMMA
    {
      const int arf = lane & 15;
      const int ah = lane >> 4;
      FragU af[4], bfr[2];
      for (int mi = 0; mi < 4; ++mi) {
        int base = (wm + mi * 16 + arf) * 40 + ah * 8;  // A: lane half -> K {0,8}
        af[mi].q[0] = *(const uint4*)(&ldsA[base]);
        af[mi].q[1] = *(const uint4*)(&ldsA[base + 16]); // +K16
      }
      for (int ni = 0; ni < 2; ++ni) {
        int base = (wn + ni * 16 + arf) * 40 + ah * 8;  // B: lane half -> K {0,8}
        bfr[ni].q[0] = *(const uint4*)(&ldsB[base]);
        bfr[ni].q[1] = *(const uint4*)(&ldsB[base + 16]); // +K16
      }
      for (int mi = 0; mi < 4; ++mi)
        for (int ni = 0; ni < 2; ++ni)
          acc[mi][ni] = __builtin_amdgcn_wmma_f32_16x16x32_bf16(
              false, af[mi].v, false, bfr[ni].v, (short)0, acc[mi][ni], false, false);
    }
  }

  // ---- epilogue / store (C layout: VGPR r -> M = r + 8*(lane>>4), N = lane&15)
  const int arf = lane & 15;
  const int ah = lane >> 4;
  for (int mi = 0; mi < 4; ++mi) {
    for (int ni = 0; ni < 2; ++ni) {
      for (int r = 0; r < 8; ++r) {
        int rowL = rowbase + wm + mi * 16 + ah * 8 + r;
        int col = colbase + wn + ni * 16 + arf;
        if (rowL < Mloc && col < Nd) {
          float v = acc[mi][ni][r] * scale;
          if (CAUSAL) { if (col > rowL) v = -1.0e30f; }
          long long crow = GROUPED ? (long long)(goff + rowL) : (long long)rowL;
          long long idx = coff + crow * (long long)ldc + col;
          if (EPI == 1) v += Res[idx];
          C[idx] = (CT)v;
        }
      }
    }
  }
}

// ---------------------------------------------------------------------------
// Small helper kernels
// ---------------------------------------------------------------------------
__global__ void rmsnorm_bf16_k(const float* __restrict__ x, const float* __restrict__ w,
                               bf16* __restrict__ out) {
  const int row = blockIdx.x;
  const int tid = threadIdx.x;
  const float* xr = x + (long long)row * D_;
  __shared__ float red[128];
  float s = 0.0f;
  for (int j = tid; j < D_; j += 128) { float v = xr[j]; s += v * v; }
  red[tid] = s; __syncthreads();
  for (int st = 64; st > 0; st >>= 1) { if (tid < st) red[tid] += red[tid + st]; __syncthreads(); }
  float inv = rsqrtf(red[0] / (float)D_ + 1e-6f);
  bf16* orow = out + (long long)row * D_;
  for (int j = tid; j < D_; j += 128) orow[j] = (bf16)(xr[j] * inv * w[j]);
}

__global__ void cvt_bf16_k(const float* __restrict__ in, bf16* __restrict__ out, int n) {
  int i = blockIdx.x * 256 + threadIdx.x;
  if (i < n) out[i] = (bf16)in[i];
}

__global__ void softmax_rows_k(const float* __restrict__ S, bf16* __restrict__ P) {
  const long long row = blockIdx.x;
  const int tid = threadIdx.x;
  const float* sr = S + row * T_;
  __shared__ float red[128];
  float m = -3.4e38f;
  for (int j = tid; j < T_; j += 128) m = fmaxf(m, sr[j]);
  red[tid] = m; __syncthreads();
  for (int st = 64; st > 0; st >>= 1) { if (tid < st) red[tid] = fmaxf(red[tid], red[tid + st]); __syncthreads(); }
  float mx = red[0]; __syncthreads();
  float s = 0.0f;
  for (int j = tid; j < T_; j += 128) s += __expf(sr[j] - mx);
  red[tid] = s; __syncthreads();
  for (int st = 64; st > 0; st >>= 1) { if (tid < st) red[tid] += red[tid + st]; __syncthreads(); }
  float inv = 1.0f / red[0];
  bf16* pr = P + row * T_;
  for (int j = tid; j < T_; j += 128) pr[j] = (bf16)(__expf(sr[j] - mx) * inv);
}

__global__ void moe_init_k(int* counts, int* curs) {
  int i = threadIdx.x;
  if (i < E_) { counts[i] = 0; curs[i] = 0; }
}

__global__ void route_topk_k(const float* __restrict__ logits, int* __restrict__ tidx,
                             float* __restrict__ gate, int* __restrict__ counts) {
  int t = blockIdx.x * 128 + threadIdx.x;
  if (t >= N_) return;
  const float* l = logits + (long long)t * E_;
  float v0 = -3.4e38f, v1 = -3.4e38f;
  int i0 = 0, i1 = 0;
  for (int e = 0; e < E_; ++e) {   // jax top_k tie-break: earlier index wins
    float v = l[e];
    if (v > v0) { v1 = v0; i1 = i0; v0 = v; i0 = e; }
    else if (v > v1) { v1 = v; i1 = e; }
  }
  float e1 = __expf(v1 - v0);
  float g0 = 1.0f / (1.0f + e1);
  tidx[2 * t] = i0; tidx[2 * t + 1] = i1;
  gate[2 * t] = g0; gate[2 * t + 1] = e1 * g0;
  atomicAdd(&counts[i0], 1);
  atomicAdd(&counts[i1], 1);
}

__global__ void moe_scan_k(const int* __restrict__ counts, int* __restrict__ offs) {
  if (threadIdx.x == 0) {
    int s = 0;
    for (int e = 0; e < E_; ++e) { offs[e] = s; s += counts[e]; }
    offs[E_] = s;
  }
}

__global__ void moe_place_k(const int* __restrict__ tidx, const int* __restrict__ offs,
                            int* __restrict__ curs, int* __restrict__ perm,
                            int* __restrict__ ppos) {
  int t = blockIdx.x * 128 + threadIdx.x;
  if (t >= N_) return;
  for (int s = 0; s < 2; ++s) {
    int e = tidx[2 * t + s];
    int pos = atomicAdd(&curs[e], 1);
    int g = offs[e] + pos;
    perm[g] = t;
    ppos[2 * t + s] = g;
  }
}

__global__ void silu_mul_k(const bf16* __restrict__ h, const bf16* __restrict__ u,
                           bf16* __restrict__ act) {
  long long g = blockIdx.x;
  int c = blockIdx.y * 256 + threadIdx.x;
  long long i = g * HID_ + c;
  float hv = (float)h[i];
  float uv = (float)u[i];
  act[i] = (bf16)((hv / (1.0f + __expf(-hv))) * uv);
}

__global__ void final_add_k(const float* __restrict__ x2, const float* __restrict__ pout,
                            const float* __restrict__ gate, const int* __restrict__ ppos,
                            float* __restrict__ out) {
  const int t = blockIdx.x;
  const int tid = threadIdx.x;
  int p0 = ppos[2 * t], p1 = ppos[2 * t + 1];
  float g0 = gate[2 * t], g1 = gate[2 * t + 1];
  const float* a = x2 + (long long)t * D_;
  const float* b0 = pout + (long long)p0 * D_;
  const float* b1 = pout + (long long)p1 * D_;
  float* o = out + (long long)t * D_;
  for (int j = tid; j < D_; j += 128) o[j] = a[j] + g0 * b0[j] + g1 * b1[j];
}

// ---------------------------------------------------------------------------
extern "C" void kernel_launch(void* const* d_in, const int* in_sizes, int n_in,
                              void* d_out, int out_size, void* d_ws, size_t ws_size,
                              hipStream_t stream) {
  (void)in_sizes; (void)n_in; (void)out_size; (void)ws_size;
  const float* x        = (const float*)d_in[0];
  const float* enc      = (const float*)d_in[1];
  const float* n1w      = (const float*)d_in[3];
  const float* n2w      = (const float*)d_in[4];
  const float* n3w      = (const float*)d_in[5];
  const float* sa_wq    = (const float*)d_in[6];
  const float* sa_wk    = (const float*)d_in[7];
  const float* sa_wv    = (const float*)d_in[8];
  const float* sa_wo    = (const float*)d_in[9];
  const float* ca_wq    = (const float*)d_in[10];
  const float* ca_wk    = (const float*)d_in[11];
  const float* ca_wv    = (const float*)d_in[12];
  const float* ca_wo    = (const float*)d_in[13];
  const float* router_w = (const float*)d_in[14];
  const float* moe_wg   = (const float*)d_in[15];
  const float* moe_wu   = (const float*)d_in[16];
  const float* moe_wd   = (const float*)d_in[17];
  float* out = (float*)d_out;

  char* ws = (char*)d_ws;
  const size_t MB = 1024 * 1024;
  bf16*  nbf    = (bf16*)(ws + 0);         // 2 MB (reused n1/n2/n3)
  bf16*  encbf  = (bf16*)(ws + 2 * MB);    // 2 MB
  float* Q      = (float*)(ws + 4 * MB);   // 4 MB
  float* Kb     = (float*)(ws + 8 * MB);   // 4 MB
  float* V      = (float*)(ws + 12 * MB);  // 4 MB
  float* Sc     = (float*)(ws + 16 * MB);  // 32 MB scores
  bf16*  Pbf    = (bf16*)(ws + 48 * MB);   // 16 MB softmax probs (bf16)
  float* attn   = (float*)(ws + 64 * MB);  // 4 MB
  float* x1     = (float*)(ws + 68 * MB);  // 4 MB
  float* x2     = (float*)(ws + 72 * MB);  // 4 MB
  float* logits = (float*)(ws + 76 * MB);  // 192 KB
  int*   tidx   = (int*)(ws + 77 * MB);
  float* gate   = (float*)(ws + 77 * MB + 64 * 1024);
  int*   counts = (int*)(ws + 77 * MB + 128 * 1024);
  int*   offs   = (int*)(ws + 77 * MB + 129 * 1024);
  int*   curs   = (int*)(ws + 77 * MB + 130 * 1024);
  int*   perm   = (int*)(ws + 77 * MB + 131 * 1024);
  int*   ppos   = (int*)(ws + 77 * MB + 160 * 1024);
  // MoE-phase aliases (attention scratch is dead by then)
  bf16*  hbuf   = (bf16*)(ws + 16 * MB);   // 16 MB (alias Sc lo)
  bf16*  ubuf   = (bf16*)(ws + 32 * MB);   // 16 MB (alias Sc hi)
  bf16*  actb   = (bf16*)(ws + 48 * MB);   // 16 MB (alias Pbf)
  float* pout   = (float*)(ws + 4 * MB);   //  8 MB (alias Q+K)

  const dim3 blk(256);
  const long long TT = (long long)T_ * T_;
  const long long TD = (long long)T_ * D_;

  moe_init_k<<<1, 64, 0, stream>>>(counts, curs);

  // ---------------- self-attention ----------------
  rmsnorm_bf16_k<<<N_, 128, 0, stream>>>(x, n1w, nbf);
  cvt_bf16_k<<<(N_ * D_) / 256, 256, 0, stream>>>(enc, encbf, N_ * D_);

  gemm_wmma<1, 0, 0, 0, 0, float><<<dim3(4, 16, 1), blk, 0, stream>>>(
      nbf, sa_wq, Q, nullptr, N_, D_, D_, D_, D_, D_,
      0, 0, 0, 0, 0, 0, 1, 1.0f, nullptr, nullptr, nullptr);
  gemm_wmma<1, 0, 0, 0, 0, float><<<dim3(4, 16, 1), blk, 0, stream>>>(
      nbf, sa_wk, Kb, nullptr, N_, D_, D_, D_, D_, D_,
      0, 0, 0, 0, 0, 0, 1, 1.0f, nullptr, nullptr, nullptr);
  gemm_wmma<1, 0, 0, 0, 0, float><<<dim3(4, 16, 1), blk, 0, stream>>>(
      nbf, sa_wv, V, nullptr, N_, D_, D_, D_, D_, D_,
      0, 0, 0, 0, 0, 0, 1, 1.0f, nullptr, nullptr, nullptr);

  gemm_wmma<0, 1, 1, 0, 0, float><<<dim3(4, 4, BH_), blk, 0, stream>>>(   // S = Q K^T (causal)
      Q, Kb, Sc, nullptr, T_, T_, DH_, D_, D_, T_,
      TD, 64, TD, 64, (long long)H_ * TT, TT, H_, 0.125f, nullptr, nullptr, nullptr);
  softmax_rows_k<<<BH_ * T_, 128, 0, stream>>>(Sc, Pbf);
  gemm_wmma<1, 0, 0, 0, 0, float><<<dim3(1, 4, BH_), blk, 0, stream>>>(   // O = P V
      Pbf, V, attn, nullptr, T_, DH_, T_, T_, D_, D_,
      (long long)H_ * TT, TT, TD, 64, TD, 64, H_, 1.0f, nullptr, nullptr, nullptr);
  gemm_wmma<0, 0, 0, 1, 0, float><<<dim3(4, 16, 1), blk, 0, stream>>>(    // x1 = x + O wo
      attn, sa_wo, x1, x, N_, D_, D_, D_, D_, D_,
      0, 0, 0, 0, 0, 0, 1, 1.0f, nullptr, nullptr, nullptr);

  // ---------------- cross-attention ----------------
  rmsnorm_bf16_k<<<N_, 128, 0, stream>>>(x1, n2w, nbf);
  gemm_wmma<1, 0, 0, 0, 0, float><<<dim3(4, 16, 1), blk, 0, stream>>>(
      nbf, ca_wq, Q, nullptr, N_, D_, D_, D_, D_, D_,
      0, 0, 0, 0, 0, 0, 1, 1.0f, nullptr, nullptr, nullptr);
  gemm_wmma<1, 0, 0, 0, 0, float><<<dim3(4, 16, 1), blk, 0, stream>>>(
      encbf, ca_wk, Kb, nullptr, N_, D_, D_, D_, D_, D_,
      0, 0, 0, 0, 0, 0, 1, 1.0f, nullptr, nullptr, nullptr);
  gemm_wmma<1, 0, 0, 0, 0, float><<<dim3(4, 16, 1), blk, 0, stream>>>(
      encbf, ca_wv, V, nullptr, N_, D_, D_, D_, D_, D_,
      0, 0, 0, 0, 0, 0, 1, 1.0f, nullptr, nullptr, nullptr);

  gemm_wmma<0, 1, 0, 0, 0, float><<<dim3(4, 4, BH_), blk, 0, stream>>>(   // no mask
      Q, Kb, Sc, nullptr, T_, T_, DH_, D_, D_, T_,
      TD, 64, TD, 64, (long long)H_ * TT, TT, H_, 0.125f, nullptr, nullptr, nullptr);
  softmax_rows_k<<<BH_ * T_, 128, 0, stream>>>(Sc, Pbf);
  gemm_wmma<1, 0, 0, 0, 0, float><<<dim3(1, 4, BH_), blk, 0, stream>>>(
      Pbf, V, attn, nullptr, T_, DH_, T_, T_, D_, D_,
      (long long)H_ * TT, TT, TD, 64, TD, 64, H_, 1.0f, nullptr, nullptr, nullptr);
  gemm_wmma<0, 0, 0, 1, 0, float><<<dim3(4, 16, 1), blk, 0, stream>>>(    // x2 = x1 + O wo
      attn, ca_wo, x2, x1, N_, D_, D_, D_, D_, D_,
      0, 0, 0, 0, 0, 0, 1, 1.0f, nullptr, nullptr, nullptr);

  // ---------------- MoE (top-2 sparse grouped GEMM) ----------------
  rmsnorm_bf16_k<<<N_, 128, 0, stream>>>(x2, n3w, nbf);
  gemm_wmma<1, 0, 0, 0, 0, float><<<dim3(1, 16, 1), blk, 0, stream>>>(    // router logits
      nbf, router_w, logits, nullptr, N_, E_, D_, D_, E_, E_,
      0, 0, 0, 0, 0, 0, 1, 1.0f, nullptr, nullptr, nullptr);
  route_topk_k<<<N_ / 128, 128, 0, stream>>>(logits, tidx, gate, counts);
  moe_scan_k<<<1, 32, 0, stream>>>(counts, offs);
  moe_place_k<<<N_ / 128, 128, 0, stream>>>(tidx, offs, curs, perm, ppos);

  gemm_wmma<1, 0, 0, 0, 1, bf16><<<dim3(16, 16, E_), blk, 0, stream>>>(   // h = X Wg[e]
      nbf, moe_wg, hbuf, nullptr, N_, HID_, D_, D_, HID_, HID_,
      0, 0, (long long)D_ * HID_, 0, 0, 0, 1, 1.0f, counts, offs, perm);
  gemm_wmma<1, 0, 0, 0, 1, bf16><<<dim3(16, 16, E_), blk, 0, stream>>>(   // u = X Wu[e]
      nbf, moe_wu, ubuf, nullptr, N_, HID_, D_, D_, HID_, HID_,
      0, 0, (long long)D_ * HID_, 0, 0, 0, 1, 1.0f, counts, offs, perm);
  silu_mul_k<<<dim3(NP_, HID_ / 256), 256, 0, stream>>>(hbuf, ubuf, actb);
  gemm_wmma<1, 0, 0, 0, 1, float><<<dim3(4, 16, E_), blk, 0, stream>>>(   // y = act Wd[e]
      actb, moe_wd, pout, nullptr, N_, D_, HID_, HID_, D_, D_,
      0, 0, (long long)HID_ * D_, 0, 0, 0, 1, 1.0f, counts, offs, nullptr);

  final_add_k<<<N_, 128, 0, stream>>>(x2, pout, gate, ppos, out);
}
